// Attention_36936718746109
// MI455X (gfx1250) — matrix operand order repaired and checked by reading, not compiled, not run on hardware
//
#include <hip/hip_runtime.h>
#include <hip/hip_bf16.h>
#include <cstdint>

// ---------------------------------------------------------------------------
// Types for CDNA5 WMMA (wave32): v_wmma_f32_16x16x32_bf16
// ---------------------------------------------------------------------------
typedef __bf16 v16bf __attribute__((ext_vector_type(16)));
typedef __bf16 bf16x8 __attribute__((ext_vector_type(8)));
typedef float  v8f   __attribute__((ext_vector_type(8)));

union FragU { v16bf v; struct { bf16x8 lo, hi; } s; };

__device__ __forceinline__ v8f wmma_bf16(v16bf a, v16bf b, v8f c) {
  // 8 args: (neg_a, A, neg_b, B, c_mod, C, reuse_a, reuse_b)
  return __builtin_amdgcn_wmma_f32_16x16x32_bf16(false, a, false, b, (short)0, c,
                                                 false, false);
}

// A-operand (16-bit 16x32, row-major source, stride ldm elements):
//   lanes 0-15 : row=lane, K = {0..7} in V0-3, {16..23} in V4-7
//   lanes 16-31: row=lane-16, K = {8..15}, {24..31}
__device__ __forceinline__ v16bf load_fragA_g(const __bf16* tile, int ldm, int lane) {
  const int r  = lane & 15;
  const int ko = (lane & 16) ? 8 : 0;
  const __bf16* p = tile + (size_t)r * ldm + ko;
  FragU f;
  f.s.lo = *reinterpret_cast<const bf16x8*>(p);
  f.s.hi = *reinterpret_cast<const bf16x8*>(p + 16);
  return f.v;
}

// A-operand built from an fp32 LDS tile (convert to bf16 on the fly)
__device__ __forceinline__ v16bf load_fragA_lds_f32(const float* tile, int ldm, int lane) {
  const int r  = lane & 15;
  const int ko = (lane & 16) ? 8 : 0;
  const float* p = tile + r * ldm + ko;
  FragU f;
#pragma unroll
  for (int i = 0; i < 8; ++i) f.s.lo[i] = (__bf16)p[i];
#pragma unroll
  for (int i = 0; i < 8; ++i) f.s.hi[i] = (__bf16)p[16 + i];
  return f.v;
}

// B-operand (16-bit 32x16). Source is "K-contiguous per output column" i.e. a
// row-major matrix whose rows are the output columns (W[n, k], K^T, V^T):
//   lanes 0-15 : col=lane,  K = {0..7} in V0-3, {16..23} in V4-7
//   lanes 16-31: col=lane-16, K = {8..15}, {24..31}
__device__ __forceinline__ v16bf load_fragB_g(const __bf16* tile, int ldn, int lane) {
  const int n  = lane & 15;
  const int ko = (lane & 16) ? 8 : 0;
  const __bf16* p = tile + (size_t)n * ldn + ko;
  FragU f;
  f.s.lo = *reinterpret_cast<const bf16x8*>(p);
  f.s.hi = *reinterpret_cast<const bf16x8*>(p + 16);
  return f.v;
}

// ---------------------------------------------------------------------------
// Kernel 0: fp32 -> bf16 conversion (grid-stride)
// ---------------------------------------------------------------------------
__global__ void attn_cvt_bf16(const float* __restrict__ s, __bf16* __restrict__ d,
                              size_t n) {
  size_t i  = (size_t)blockIdx.x * blockDim.x + threadIdx.x;
  size_t st = (size_t)gridDim.x * blockDim.x;
  for (; i < n; i += st) d[i] = (__bf16)s[i];
}

// ---------------------------------------------------------------------------
// Kernel 1: fused QKV projection, 2x2 register blocking.
//   One wave = 32 rows x 32 cols of Q, K and V (12 WMMA accumulators).
//   Q,K -> [BH, N, 64] bf16 ; V -> transposed [BH, 64, N] bf16
// ---------------------------------------------------------------------------
__global__ void __launch_bounds__(256) attn_qkv(
    const __bf16* __restrict__ xb, const __bf16* __restrict__ wq,
    const __bf16* __restrict__ wk, const __bf16* __restrict__ wv,
    const float* __restrict__ bq, const float* __restrict__ bk,
    const float* __restrict__ bv,
    __bf16* __restrict__ Qh, __bf16* __restrict__ Kh, __bf16* __restrict__ Vt) {
  const int lane = threadIdx.x & 31;
  const int gw   = blockIdx.x * 8 + (threadIdx.x >> 5);
  const int mt   = gw >> 5;              // 0..1023 : 32-row block of x
  const int ct   = gw & 31;              // 0..31   : 32-col block of channels
  const int m0   = mt << 5;
  const int c0   = ct << 5;

  v8f aq[2][2] = {}, ak[2][2] = {}, av[2][2] = {};
  const __bf16* xr0 = xb + (size_t)m0 * 1024;
  const __bf16* xr1 = xb + (size_t)(m0 + 16) * 1024;
  const __bf16* wqr0 = wq + (size_t)c0 * 1024;
  const __bf16* wqr1 = wq + (size_t)(c0 + 16) * 1024;
  const __bf16* wkr0 = wk + (size_t)c0 * 1024;
  const __bf16* wkr1 = wk + (size_t)(c0 + 16) * 1024;
  const __bf16* wvr0 = wv + (size_t)c0 * 1024;
  const __bf16* wvr1 = wv + (size_t)(c0 + 16) * 1024;

  for (int k = 0; k < 1024; k += 32) {
    const v16bf a0 = load_fragA_g(xr0 + k, 1024, lane);
    const v16bf a1 = load_fragA_g(xr1 + k, 1024, lane);
    {
      const v16bf b0 = load_fragB_g(wqr0 + k, 1024, lane);
      const v16bf b1 = load_fragB_g(wqr1 + k, 1024, lane);
      aq[0][0] = wmma_bf16(a0, b0, aq[0][0]);
      aq[0][1] = wmma_bf16(a0, b1, aq[0][1]);
      aq[1][0] = wmma_bf16(a1, b0, aq[1][0]);
      aq[1][1] = wmma_bf16(a1, b1, aq[1][1]);
    }
    {
      const v16bf b0 = load_fragB_g(wkr0 + k, 1024, lane);
      const v16bf b1 = load_fragB_g(wkr1 + k, 1024, lane);
      ak[0][0] = wmma_bf16(a0, b0, ak[0][0]);
      ak[0][1] = wmma_bf16(a0, b1, ak[0][1]);
      ak[1][0] = wmma_bf16(a1, b0, ak[1][0]);
      ak[1][1] = wmma_bf16(a1, b1, ak[1][1]);
    }
    {
      const v16bf b0 = load_fragB_g(wvr0 + k, 1024, lane);
      const v16bf b1 = load_fragB_g(wvr1 + k, 1024, lane);
      av[0][0] = wmma_bf16(a0, b0, av[0][0]);
      av[0][1] = wmma_bf16(a0, b1, av[0][1]);
      av[1][0] = wmma_bf16(a1, b0, av[1][0]);
      av[1][1] = wmma_bf16(a1, b1, av[1][1]);
    }
  }

  const int col   = lane & 15;
  const int rbase = (lane & 16) ? 8 : 0;
  const int b     = m0 >> 9;             // 32-row block never crosses a batch
#pragma unroll
  for (int i = 0; i < 2; ++i) {
#pragma unroll
    for (int j = 0; j < 2; ++j) {
      const int c  = c0 + j * 16 + col;
      const int h  = c >> 6;
      const int dk = c & 63;
      const int bh = b * 16 + h;
      const int nbase = ((m0 + i * 16) & 511) + rbase;
      const float fq = bq[c], fk = bk[c], fv = bv[c];
#pragma unroll
      for (int v = 0; v < 8; ++v) {
        const int n = nbase + v;
        Qh[((size_t)bh * 512 + n) * 64 + dk] = (__bf16)(aq[i][j][v] + fq);
        Kh[((size_t)bh * 512 + n) * 64 + dk] = (__bf16)(ak[i][j][v] + fk);
        Vt[((size_t)bh * 64 + dk) * 512 + n] = (__bf16)(av[i][j][v] + fv);
      }
    }
  }
}

// ---------------------------------------------------------------------------
// Kernel 2: attention core. One wave (one block) per (head, 16-row block).
//   scores -> fp32 LDS [16 x 516] -> softmax -> probs (global fp32)
//   ctx = P @ V via WMMA (A from LDS, B from V^T), stored bf16 [B, N, C]
// ---------------------------------------------------------------------------
__global__ void __launch_bounds__(32) attn_core(
    const __bf16* __restrict__ Qh, const __bf16* __restrict__ Kh,
    const __bf16* __restrict__ Vt, float* __restrict__ probs,
    __bf16* __restrict__ ctx) {
  __shared__ float S[16 * 516];          // 33 KB, stride padded (bank spread, 16B aligned)
  const int lane  = threadIdx.x & 31;
  const int bh    = blockIdx.x >> 5;     // 0..1023
  const int i0    = (blockIdx.x & 31) << 4;
  const int col   = lane & 15;
  const int rbase = (lane & 16) ? 8 : 0;

  // Q tile (rows i0..i0+15, full d=64) as two A fragments, reused for all K tiles
  const __bf16* qtile = Qh + ((size_t)bh * 512 + i0) * 64;
  const v16bf a0 = load_fragA_g(qtile, 64, lane);
  const v16bf a1 = load_fragA_g(qtile + 32, 64, lane);
  const float scale = 0.125f;            // 1/sqrt(64)

  // ---- scores = Q K^T * scale -> LDS
  for (int jt = 0; jt < 32; ++jt) {
    const __bf16* ktile = Kh + ((size_t)bh * 512 + jt * 16) * 64;
    v8f acc = {};
    acc = wmma_bf16(a0, load_fragB_g(ktile, 64, lane), acc);
    acc = wmma_bf16(a1, load_fragB_g(ktile + 32, 64, lane), acc);
#pragma unroll
    for (int v = 0; v < 8; ++v)
      S[(v + rbase) * 516 + jt * 16 + col] = acc[v] * scale;
  }
  __syncthreads();

  // ---- softmax: lane&15 -> row, lane bit4 selects column half; pair via shfl_xor(16)
  {
    const int row = lane & 15;
    const int cb  = (lane & 16) ? 256 : 0;
    float* Srow = &S[row * 516 + cb];
    float m = -3.0e38f;
    for (int ci = 0; ci < 256; ++ci) m = fmaxf(m, Srow[ci]);
    m = fmaxf(m, __shfl_xor(m, 16, 32));
    float ssum = 0.f;
    for (int ci = 0; ci < 256; ++ci) ssum += __expf(Srow[ci] - m);
    ssum += __shfl_xor(ssum, 16, 32);
    const float inv = 1.0f / ssum;
    for (int ci = 0; ci < 256; ++ci) Srow[ci] = __expf(Srow[ci] - m) * inv;
  }
  __syncthreads();

  // ---- coalesced probs dump (fp32, [BH, N, N])
  {
    float* prow = probs + ((size_t)bh * 512 + i0) * 512;
    for (int r = 0; r < 16; ++r)
      for (int ci = lane; ci < 512; ci += 32)
        prow[(size_t)r * 512 + ci] = S[r * 516 + ci];
  }

  // ---- ctx = P @ V : A from LDS (cvt f32->bf16), B from V^T rows
  v8f c0v = {}, c1v = {}, c2v = {}, c3v = {};
  for (int kt = 0; kt < 16; ++kt) {
    v16bf a = load_fragA_lds_f32(&S[kt * 32], 516, lane);
    const __bf16* vb = Vt + (size_t)bh * 64 * 512 + kt * 32;
    c0v = wmma_bf16(a, load_fragB_g(vb + (size_t)0  * 512, 512, lane), c0v);
    c1v = wmma_bf16(a, load_fragB_g(vb + (size_t)16 * 512, 512, lane), c1v);
    c2v = wmma_bf16(a, load_fragB_g(vb + (size_t)32 * 512, 512, lane), c2v);
    c3v = wmma_bf16(a, load_fragB_g(vb + (size_t)48 * 512, 512, lane), c3v);
  }
  // merge heads: ctx layout [B, N, C] with C index = h*64 + dk
  const int b = bh >> 4, h = bh & 15;
  __bf16* crow = ctx + ((size_t)b * 512 + i0 + rbase) * 1024 + h * 64 + col;
#pragma unroll
  for (int v = 0; v < 8; ++v) {
    crow[(size_t)v * 1024 + 0]  = (__bf16)c0v[v];
    crow[(size_t)v * 1024 + 16] = (__bf16)c1v[v];
    crow[(size_t)v * 1024 + 32] = (__bf16)c2v[v];
    crow[(size_t)v * 1024 + 48] = (__bf16)c3v[v];
  }
}

// ---------------------------------------------------------------------------
// Kernel 3: out = ctx @ Wo^T + bo   (M=32768, N=512, K=1024), fp32 output.
//   4x4 register blocking: one wave = 64x64 output, 16 accumulators.
// ---------------------------------------------------------------------------
__global__ void __launch_bounds__(256) attn_out(
    const __bf16* __restrict__ ctx, const __bf16* __restrict__ wo,
    const float* __restrict__ bo, float* __restrict__ out) {
  const int lane = threadIdx.x & 31;
  const int gw   = blockIdx.x * 8 + (threadIdx.x >> 5);
  const int mt   = gw >> 3;              // 0..511 : 64-row block
  const int nt   = gw & 7;               // 0..7   : 64-col block
  const int m0   = mt << 6, n0 = nt << 6;

  v8f acc[4][4] = {};
  for (int k = 0; k < 1024; k += 32) {
    v16bf a[4];
#pragma unroll
    for (int i = 0; i < 4; ++i)
      a[i] = load_fragA_g(ctx + (size_t)(m0 + 16 * i) * 1024 + k, 1024, lane);
#pragma unroll
    for (int j = 0; j < 4; ++j) {
      const v16bf bfr = load_fragB_g(wo + (size_t)(n0 + 16 * j) * 1024 + k, 1024, lane);
#pragma unroll
      for (int i = 0; i < 4; ++i) acc[i][j] = wmma_bf16(a[i], bfr, acc[i][j]);
    }
  }

  const int col   = lane & 15;
  const int rbase = (lane & 16) ? 8 : 0;
#pragma unroll
  for (int j = 0; j < 4; ++j) {
    const float bias = bo[n0 + 16 * j + col];
#pragma unroll
    for (int i = 0; i < 4; ++i) {
      float* orow = out + (size_t)(m0 + 16 * i + rbase) * 512 + n0 + 16 * j + col;
#pragma unroll
      for (int v = 0; v < 8; ++v) orow[(size_t)v * 512] = acc[i][j][v] + bias;
    }
  }
}

// ---------------------------------------------------------------------------
// Launch
// ---------------------------------------------------------------------------
extern "C" void kernel_launch(void* const* d_in, const int* in_sizes, int n_in,
                              void* d_out, int out_size, void* d_ws, size_t ws_size,
                              hipStream_t stream) {
  const float* x  = (const float*)d_in[0];
  const float* Wq = (const float*)d_in[1];
  const float* bq = (const float*)d_in[2];
  const float* Wk = (const float*)d_in[3];
  const float* bk = (const float*)d_in[4];
  const float* Wv = (const float*)d_in[5];
  const float* bv = (const float*)d_in[6];
  const float* Wo = (const float*)d_in[7];
  const float* bo = (const float*)d_in[8];

  float* out   = (float*)d_out;                       // [64,512,512]
  float* probs = out + (size_t)64 * 512 * 512;        // [1024,512,512]

  // workspace carve-out (~327 MB total)
  char* ws = (char*)d_ws;
  size_t off = 0;
  auto carve = [&](size_t elems) {
    __bf16* p = (__bf16*)(ws + off);
    off += elems * sizeof(__bf16);
    return p;
  };
  const size_t NX = (size_t)32768 * 1024;   // x / Q / K / V / ctx element count
  __bf16* xb  = carve(NX);
  __bf16* wq  = carve((size_t)1024 * 1024);
  __bf16* wk  = carve((size_t)1024 * 1024);
  __bf16* wv  = carve((size_t)1024 * 1024);
  __bf16* wo  = carve((size_t)512 * 1024);
  __bf16* Qh  = carve(NX);                  // [1024, 512, 64]
  __bf16* Kh  = carve(NX);                  // [1024, 512, 64]
  __bf16* Vt  = carve(NX);                  // [1024, 64, 512]
  __bf16* ctxp = carve(NX);                 // [64, 512, 1024]
  (void)ws_size; (void)in_sizes; (void)n_in; (void)out_size;

  // 0) bf16 conversions
  attn_cvt_bf16<<<2048, 256, 0, stream>>>(x, xb, NX);
  attn_cvt_bf16<<<512, 256, 0, stream>>>(Wq, wq, (size_t)1024 * 1024);
  attn_cvt_bf16<<<512, 256, 0, stream>>>(Wk, wk, (size_t)1024 * 1024);
  attn_cvt_bf16<<<512, 256, 0, stream>>>(Wv, wv, (size_t)1024 * 1024);
  attn_cvt_bf16<<<256, 256, 0, stream>>>(Wo, wo, (size_t)512 * 1024);

  // 1) QKV projections: 1024 x 32 blocks of 32x32, 8 waves / block
  attn_qkv<<<4096, 256, 0, stream>>>(xb, wq, wk, wv, bq, bk, bv, Qh, Kh, Vt);

  // 2) attention core: 1024 heads x 32 row-blocks, one wave per block
  attn_core<<<32768, 32, 0, stream>>>(Qh, Kh, Vt, probs, ctxp);

  // 3) output GEMM: 512 x 8 blocks of 64x64, 8 waves / block
  attn_out<<<512, 256, 0, stream>>>(ctxp, wo, bo, out);
}
